// LGF_50663434224442
// MI455X (gfx1250) — compile-verified, matrix operand order checked by reading, not run on hardware
//
#include <hip/hip_runtime.h>

// ---------------------------------------------------------------------------
// Types / WMMA helpers (CDNA5 gfx1250, wave32)
// ---------------------------------------------------------------------------
typedef __bf16 bf16;
typedef __attribute__((ext_vector_type(16))) __bf16 v16bf;
typedef __attribute__((ext_vector_type(8)))  __bf16 v8bf;
typedef __attribute__((ext_vector_type(8)))  float  v8f;
typedef __attribute__((ext_vector_type(4)))  float  v4f;

#define B_N   4
#define DIM_C 128
#define L_S   4096            // H*W
#define DIN   256
#define NST   16
#define NROW  16384           // B_N * L_S

__device__ __forceinline__ v8f wmma_bf16(v16bf a, v16bf b, v8f c) {
    // D = A(16x32 bf16) x B(32x16 bf16) + C(16x16 f32)
    return __builtin_amdgcn_wmma_f32_16x16x32_bf16(
        /*neg_a=*/false, a, /*neg_b=*/false, b,
        /*c_mod=*/(short)0, c, /*reuse_a=*/false, /*reuse_b=*/false);
}

// Build one lane's 16-element fragment from a bf16 row pointer (row-major, K
// contiguous). ISA layout: lane khalf=lane>>4 holds K = 8*khalf+0..7 in
// elements 0..7 and K = 16+8*khalf+0..7 in elements 8..15.  -> two 16B loads.
__device__ __forceinline__ v16bf load_frag(const bf16* rowp, int k0, int khalf) {
    v8bf lo = *(const v8bf*)(rowp + k0 + 8 * khalf);
    v8bf hi = *(const v8bf*)(rowp + k0 + 16 + 8 * khalf);
    v16bf r;
#pragma unroll
    for (int j = 0; j < 8; ++j) { r[j] = lo[j]; r[j + 8] = hi[j]; }
    return r;
}

// ---------------------------------------------------------------------------
// k_prep: weight conversion fp32 -> bf16, xpw zero-padded 40->48 rows,
//         A = -exp(A_log) precompute.
// ---------------------------------------------------------------------------
__global__ void k_prep(const float* __restrict__ inproj,
                       const float* __restrict__ xpw, const float* __restrict__ xpwb,
                       const float* __restrict__ op1, const float* __restrict__ op2,
                       const float* __restrict__ alog, const float* __restrict__ alogb,
                       bf16* __restrict__ w_inproj, bf16* __restrict__ w_xp,
                       bf16* __restrict__ w_op, float* __restrict__ aneg) {
    int i = blockIdx.x * blockDim.x + threadIdx.x;          // 0 .. 65535
    if (i < DIN * DIM_C) w_inproj[i] = (bf16)inproj[i];
    if (i < 2 * 48 * DIN) {
        int dir = i / (48 * DIN);
        int r   = (i / DIN) % 48;
        int c   = i % DIN;
        const float* src = dir ? xpwb : xpw;
        w_xp[i] = (bf16)((r < 40) ? src[r * DIN + c] : 0.f);
    }
    if (i < 2 * DIM_C * DIN) {
        int dir = i / (DIM_C * DIN);
        int j   = i % (DIM_C * DIN);
        w_op[i] = (bf16)(dir ? op2[j] : op1[j]);
    }
    if (i < 2 * DIN * NST) {
        int dir = i / (DIN * NST);
        int j   = i % (DIN * NST);
        aneg[i] = -__expf(dir ? alogb[j] : alog[j]);
    }
}

// ---------------------------------------------------------------------------
// k_ln_inproj: fused LayerNorm(C=128) + GEMM  xz[m,256] = LN(x)[m,128] @ W^T
// x is (B, C, H*W) channel-major.  1 block = 16 token rows, 128 threads.
// ---------------------------------------------------------------------------
__global__ __launch_bounds__(128) void k_ln_inproj(
        const float* __restrict__ x, const float* __restrict__ lnw,
        const float* __restrict__ lnb, const bf16* __restrict__ W,
        bf16* __restrict__ xz) {
    __shared__ float sf[16 * 132];       // f32 staging tile [16][128] pad->132
    __shared__ bf16  sb[16 * 136];       // bf16 LN tile     [16][128] pad->136

    const int t  = threadIdx.x;
    const int m0 = blockIdx.x * 16;
    const int b  = m0 >> 12;             // L = 4096
    const int l0 = m0 & (L_S - 1);

    // Stage: quad-of-lanes reads 64B of consecutive tokens for one channel.
    const int q = t & 3, cb = t >> 2;    // cb in 0..31
#pragma unroll
    for (int i = 0; i < 4; ++i) {
        int c = cb + 32 * i;
        v4f v = *(const v4f*)(x + (size_t)(b * DIM_C + c) * L_S + l0 + q * 4);
#pragma unroll
        for (int j = 0; j < 4; ++j) sf[(q * 4 + j) * 132 + c] = v[j];
    }
    __syncthreads();

    // LayerNorm: 8 lanes per row.
    const int r = t >> 3, g = t & 7;
    float vals[16], s1 = 0.f, s2 = 0.f;
#pragma unroll
    for (int j = 0; j < 16; ++j) {
        float v = sf[r * 132 + g * 16 + j];
        vals[j] = v; s1 += v; s2 += v * v;
    }
#pragma unroll
    for (int m = 1; m < 8; m <<= 1) {
        s1 += __shfl_xor(s1, m, 8);
        s2 += __shfl_xor(s2, m, 8);
    }
    const float mean = s1 * (1.f / 128.f);
    const float inv  = rsqrtf(s2 * (1.f / 128.f) - mean * mean + 1e-5f);
#pragma unroll
    for (int j = 0; j < 16; ++j) {
        int c = g * 16 + j;
        sb[r * 136 + c] = (bf16)((vals[j] - mean) * inv * lnw[c] + lnb[c]);
    }
    __syncthreads();

    // WMMA GEMM: 4 waves, each covers 4 N-tiles of 16; K = 128 -> 4 chunks.
    const int wv = t >> 5, lane = t & 31;
    const int mrow = lane & 15, khalf = lane >> 4;
    v16bf afr[4];
#pragma unroll
    for (int kc = 0; kc < 4; ++kc)
        afr[kc] = load_frag(sb + mrow * 136, kc * 32, khalf);

#pragma unroll
    for (int i = 0; i < 4; ++i) {
        const int nt = wv * 4 + i;
        const int n  = nt * 16 + mrow;
        v8f acc = {};
#pragma unroll
        for (int kc = 0; kc < 4; ++kc) {
            v16bf bfr = load_frag(W + n * DIM_C, kc * 32, khalf);
            acc = wmma_bf16(afr[kc], bfr, acc);
        }
#pragma unroll
        for (int r2 = 0; r2 < 8; ++r2) {
            int mm = r2 + 8 * khalf;
            xz[(size_t)(m0 + mm) * DIN + n] = (bf16)acc[r2];
        }
    }
}

// ---------------------------------------------------------------------------
// k_conv: causal depthwise conv K=4 (fwd + bwd weight sets) + SiLU.
// xz is (B*L, 256) bf16; outputs same layout, bf16.
// ---------------------------------------------------------------------------
__global__ __launch_bounds__(256) void k_conv(
        const bf16* __restrict__ xz,
        const float* __restrict__ wf, const float* __restrict__ bfc,
        const float* __restrict__ wb, const float* __restrict__ bbc,
        bf16* __restrict__ cxf, bf16* __restrict__ cxb) {
    const int gid  = blockIdx.x * 256 + threadIdx.x;   // B*L*256 threads
    const int d    = gid & 255;
    const int rowl = gid >> 8;
    const int l    = rowl & (L_S - 1);
    float accf = bfc[d], accb = bbc[d];
#pragma unroll
    for (int k = 0; k < 4; ++k) {
        int lm  = l - 3 + k;
        float v = (lm >= 0) ? (float)xz[(size_t)(rowl - 3 + k) * DIN + d] : 0.f;
        accf += wf[d * 4 + k] * v;
        accb += wb[d * 4 + k] * v;
    }
    accf = accf / (1.f + __expf(-accf));   // SiLU
    accb = accb / (1.f + __expf(-accb));
    cxf[gid] = (bf16)accf;
    cxb[gid] = (bf16)accb;
}

// ---------------------------------------------------------------------------
// k_xproj: dbl[m,48] = u[m,256] @ W48^T   (W48 = xpw padded 40->48, bf16)
// 1 wave per block, A fragments straight from global bf16 rows.
// ---------------------------------------------------------------------------
__global__ __launch_bounds__(32) void k_xproj(
        const bf16* __restrict__ u, const bf16* __restrict__ W48,
        float* __restrict__ dbl) {
    const int m0 = blockIdx.x * 16;
    const int lane = threadIdx.x;
    const int mrow = lane & 15, khalf = lane >> 4;
    const bf16* arow = u + (size_t)(m0 + mrow) * DIN;
    v16bf afr[8];
#pragma unroll
    for (int kc = 0; kc < 8; ++kc) afr[kc] = load_frag(arow, kc * 32, khalf);

#pragma unroll
    for (int nt = 0; nt < 3; ++nt) {
        const int n = nt * 16 + mrow;
        v8f acc = {};
#pragma unroll
        for (int kc = 0; kc < 8; ++kc) {
            v16bf bfr = load_frag(W48 + n * DIN, kc * 32, khalf);
            acc = wmma_bf16(afr[kc], bfr, acc);
        }
#pragma unroll
        for (int r2 = 0; r2 < 8; ++r2)
            dbl[(size_t)(m0 + r2 + 8 * khalf) * 48 + n] = acc[r2];
    }
}

// ---------------------------------------------------------------------------
// k_scan: fused softplus(dt) + selective scan.  One launch covers all
// (input, dir, batch, channel-group) tiles -> 16 fully parallel scans.
// Thread = (d, n) pair; 16 lanes per channel. Backward dir iterates l in
// reverse (equivalent to reverse/scan/reverse in the reference).
// ---------------------------------------------------------------------------
__global__ __launch_bounds__(256) void k_scan(
        const bf16* u00, const bf16* u01, const bf16* u10, const bf16* u11,
        const float* d00, const float* d01, const float* d10, const float* d11,
        bf16* y00, bf16* y01, bf16* y10, bf16* y11,
        const float* __restrict__ aneg,
        const float* dtw, const float* dtwb,
        const float* dtb, const float* dtbb,
        const float* Dp,  const float* Dpb) {
    const int inp = blockIdx.z, dir = blockIdx.y;
    const bf16*  u    = inp == 0 ? (dir == 0 ? u00 : u01) : (dir == 0 ? u10 : u11);
    const float* dbl  = inp == 0 ? (dir == 0 ? d00 : d01) : (dir == 0 ? d10 : d11);
    bf16*        yout = inp == 0 ? (dir == 0 ? y00 : y01) : (dir == 0 ? y10 : y11);
    const float* dtw_ = dir ? dtwb : dtw;
    const float* dtb_ = dir ? dtbb : dtb;
    const float* Dp_  = dir ? Dpb  : Dp;
    const float* an   = aneg + dir * (DIN * NST);

    const int b  = blockIdx.x >> 4;
    const int dg = blockIdx.x & 15;
    const int t  = threadIdx.x;
    const int d  = dg * 16 + (t >> 4);
    const int n  = t & 15;
    const float A_n = an[d * NST + n];

    float dtwr[8], dtbd = 0.f, Dpd = 0.f;
    if (n == 0) {
#pragma unroll
        for (int rr = 0; rr < 8; ++rr) dtwr[rr] = dtw_[d * 8 + rr];
        dtbd = dtb_[d];
        Dpd  = Dp_[d];
    }

    int l = (dir == 0) ? 0 : (L_S - 1);
    const int step = (dir == 0) ? 1 : -1;
    float h = 0.f;
    for (int s = 0; s < L_S; ++s, l += step) {
        const int row = b * L_S + l;
        if (((s & 7) == 0) && n == 0 && (s + 8) < L_S)
            __builtin_prefetch(dbl + (size_t)(row + step * 8) * 48, 0, 1);

        float dt = 0.f;
        if (n == 0) {
            const float* dr = dbl + (size_t)row * 48;
            float acc = dtbd;
#pragma unroll
            for (int rr = 0; rr < 8; ++rr) acc += dr[rr] * dtwr[rr];
            dt = (acc > 20.f) ? acc : log1pf(__expf(acc));    // softplus
        }
        dt = __shfl(dt, 0, 16);

        const float Bn = dbl[(size_t)row * 48 + 8 + n];
        const float Cn = dbl[(size_t)row * 48 + 24 + n];
        const float uu = (float)u[(size_t)row * DIN + d];

        h = __expf(dt * A_n) * h + (dt * uu) * Bn;
        float y = h * Cn;
        y += __shfl_xor(y, 1, 16);
        y += __shfl_xor(y, 2, 16);
        y += __shfl_xor(y, 4, 16);
        y += __shfl_xor(y, 8, 16);
        if (n == 0) yout[(size_t)row * DIN + d] = (bf16)(y + uu * Dpd);
    }
}

// ---------------------------------------------------------------------------
// k_gate: out[b,c,l] = ((g1a+g1b) * relu(g2a+g2b))[m,256] @ W^T  (N=128)
// g*b may be nullptr (single-source gate). 1 block = 16 rows, 128 threads.
// ---------------------------------------------------------------------------
__global__ __launch_bounds__(128) void k_gate(
        const bf16* __restrict__ g1a, const bf16* __restrict__ g1b,
        const bf16* __restrict__ g2a, const bf16* __restrict__ g2b,
        const bf16* __restrict__ W, float* __restrict__ out) {
    __shared__ bf16 sg[16 * 264];        // gate tile [16][256] pad->264

    const int t  = threadIdx.x;
    const int m0 = blockIdx.x * 16;
    const int b  = m0 >> 12;
    const int l0 = m0 & (L_S - 1);

    // Stage gate tile (elementwise combine in f32 -> bf16 LDS).
    const int row = t >> 3;
    const int c0  = (t & 7) * 32;
    const size_t rb = (size_t)(m0 + row) * DIN;
#pragma unroll
    for (int ch = 0; ch < 4; ++ch) {
        const int c = c0 + ch * 8;
        float v1[8], v2[8];
        { v8bf a = *(const v8bf*)(g1a + rb + c);
#pragma unroll
          for (int j = 0; j < 8; ++j) v1[j] = (float)a[j]; }
        if (g1b) { v8bf a = *(const v8bf*)(g1b + rb + c);
#pragma unroll
          for (int j = 0; j < 8; ++j) v1[j] += (float)a[j]; }
        { v8bf a = *(const v8bf*)(g2a + rb + c);
#pragma unroll
          for (int j = 0; j < 8; ++j) v2[j] = (float)a[j]; }
        if (g2b) { v8bf a = *(const v8bf*)(g2b + rb + c);
#pragma unroll
          for (int j = 0; j < 8; ++j) v2[j] += (float)a[j]; }
#pragma unroll
        for (int j = 0; j < 8; ++j)
            sg[row * 264 + c + j] = (bf16)(v1[j] * fmaxf(v2[j], 0.f));
    }
    __syncthreads();

    // WMMA GEMM: 4 waves x 2 N-tiles each (N=128), K=256 -> 8 chunks.
    const int wv = t >> 5, lane = t & 31;
    const int mrow = lane & 15, khalf = lane >> 4;
    v16bf afr[8];
#pragma unroll
    for (int kc = 0; kc < 8; ++kc)
        afr[kc] = load_frag(sg + mrow * 264, kc * 32, khalf);

#pragma unroll
    for (int i = 0; i < 2; ++i) {
        const int nt = wv * 2 + i;
        const int n  = nt * 16 + mrow;        // output channel c
        v8f acc = {};
#pragma unroll
        for (int kc = 0; kc < 8; ++kc) {
            v16bf bfr = load_frag(W + n * DIN, kc * 32, khalf);
            acc = wmma_bf16(afr[kc], bfr, acc);
        }
        // out layout (B, 128, L): out[(b*128+c)*L + l]
#pragma unroll
        for (int r2 = 0; r2 < 8; ++r2) {
            int mm = r2 + 8 * khalf;
            out[(size_t)(b * DIM_C + n) * L_S + l0 + mm] = acc[r2];
        }
    }
}

// ---------------------------------------------------------------------------
// Host launcher
// ---------------------------------------------------------------------------
extern "C" void kernel_launch(void* const* d_in, const int* in_sizes, int n_in,
                              void* d_out, int out_size, void* d_ws, size_t ws_size,
                              hipStream_t stream) {
    const float* x1       = (const float*)d_in[0];
    const float* x2       = (const float*)d_in[1];
    const float* ln1_w    = (const float*)d_in[2];
    const float* ln1_b    = (const float*)d_in[3];
    const float* ln2_w    = (const float*)d_in[4];
    const float* ln2_b    = (const float*)d_in[5];
    const float* in_proj  = (const float*)d_in[6];
    const float* conv_w   = (const float*)d_in[7];
    const float* conv_b   = (const float*)d_in[8];
    const float* xpw      = (const float*)d_in[9];
    const float* dtw      = (const float*)d_in[10];
    const float* dtb      = (const float*)d_in[11];
    const float* A_log    = (const float*)d_in[12];
    const float* Dp       = (const float*)d_in[13];
    const float* conv_w_b = (const float*)d_in[14];
    const float* conv_b_b = (const float*)d_in[15];
    const float* xpw_b    = (const float*)d_in[16];
    const float* dtw_b    = (const float*)d_in[17];
    const float* dtb_b    = (const float*)d_in[18];
    const float* A_log_b  = (const float*)d_in[19];
    const float* Dp_b     = (const float*)d_in[20];
    const float* op1      = (const float*)d_in[21];
    const float* op2      = (const float*)d_in[22];
    float* out = (float*)d_out;

    char* ws = (char*)d_ws;
    constexpr size_t O_WINPROJ = 0;                                   // 64 KB
    constexpr size_t O_WXP  = O_WINPROJ + 256 * 128 * 2;              // 48 KB
    constexpr size_t O_WOP  = O_WXP + 2 * 48 * 256 * 2;               // 128 KB
    constexpr size_t O_ANEG = O_WOP + 2 * 128 * 256 * 2;              // 32 KB
    constexpr size_t O_XZ   = O_ANEG + 2 * 256 * 16 * 4;
    constexpr size_t SZA    = (size_t)NROW * DIN * 2;                 // bf16 act, 8 MB
    constexpr size_t O_CX   = O_XZ + SZA;
    constexpr size_t O_CXB  = O_CX + 2 * SZA;
    constexpr size_t O_DBL  = O_CXB + 2 * SZA;
    constexpr size_t SZD    = (size_t)NROW * 48 * 4;                  // f32 dbl, 3 MB
    constexpr size_t O_Y    = O_DBL + 4 * SZD;                        // + 4*8 MB y

    bf16*  w_inproj = (bf16*)(ws + O_WINPROJ);
    bf16*  w_xp     = (bf16*)(ws + O_WXP);     // [2][48][256]
    bf16*  w_op     = (bf16*)(ws + O_WOP);     // [2][128][256]
    float* aneg     = (float*)(ws + O_ANEG);   // [2][256][16]
    bf16*  xz       = (bf16*)(ws + O_XZ);
    bf16*  cx0      = (bf16*)(ws + O_CX);
    bf16*  cx1      = cx0 + (size_t)NROW * DIN;
    bf16*  cxb0     = (bf16*)(ws + O_CXB);
    bf16*  cxb1     = cxb0 + (size_t)NROW * DIN;
    float* dbl00    = (float*)(ws + O_DBL);
    float* dbl01    = dbl00 + (size_t)NROW * 48;
    float* dbl10    = dbl01 + (size_t)NROW * 48;
    float* dbl11    = dbl10 + (size_t)NROW * 48;
    bf16*  y00      = (bf16*)(ws + O_Y);
    bf16*  y01      = y00 + (size_t)NROW * DIN;
    bf16*  y10      = y01 + (size_t)NROW * DIN;
    bf16*  y11      = y10 + (size_t)NROW * DIN;

    const int MTILES = NROW / 16;                       // 1024
    const size_t OUTN = (size_t)B_N * DIM_C * L_S;      // per-output elems

    k_prep<<<256, 256, 0, stream>>>(in_proj, xpw, xpw_b, op1, op2, A_log, A_log_b,
                                    w_inproj, w_xp, w_op, aneg);

    // input 0
    k_ln_inproj<<<MTILES, 128, 0, stream>>>(x1, ln1_w, ln1_b, w_inproj, xz);
    k_conv<<<NROW, 256, 0, stream>>>(xz, conv_w, conv_b, conv_w_b, conv_b_b, cx0, cxb0);
    k_xproj<<<MTILES, 32, 0, stream>>>(cx0,  w_xp,            dbl00);
    k_xproj<<<MTILES, 32, 0, stream>>>(cxb0, w_xp + 48 * 256, dbl01);

    // input 1 (xz buffer reused)
    k_ln_inproj<<<MTILES, 128, 0, stream>>>(x2, ln2_w, ln2_b, w_inproj, xz);
    k_conv<<<NROW, 256, 0, stream>>>(xz, conv_w, conv_b, conv_w_b, conv_b_b, cx1, cxb1);
    k_xproj<<<MTILES, 32, 0, stream>>>(cx1,  w_xp,            dbl10);
    k_xproj<<<MTILES, 32, 0, stream>>>(cxb1, w_xp + 48 * 256, dbl11);

    // all 16 independent scans in one launch: grid (b*16 chan-groups, dir, input)
    k_scan<<<dim3(B_N * 16, 2, 2), 256, 0, stream>>>(
        cx0, cxb0, cx1, cxb1,
        dbl00, dbl01, dbl10, dbl11,
        y00, y01, y10, y11,
        aneg, dtw, dtw_b, dtb, dtb_b, Dp, Dp_b);

    // gated out-projections; s6_i = y_i0 + y_i1 formed inside k_gate.
    bf16* wop1 = w_op;
    bf16* wop2 = w_op + (size_t)128 * 256;
    k_gate<<<MTILES, 128, 0, stream>>>(y00, y01, y10, y11,           wop1, out + 0 * OUTN); // G1
    k_gate<<<MTILES, 128, 0, stream>>>(y00, y01, cx1, nullptr,       wop1, out + 1 * OUTN); // L1
    k_gate<<<MTILES, 128, 0, stream>>>(y10, y11, y00, y01,           wop2, out + 2 * OUTN); // G2
    k_gate<<<MTILES, 128, 0, stream>>>(y10, y11, cx0, nullptr,       wop2, out + 3 * OUTN); // L2

    (void)in_sizes; (void)n_in; (void)out_size; (void)ws_size;
}
